// TrajectoryModelingWithMotion_74105365725204
// MI455X (gfx1250) — matrix operand order repaired and checked by reading, not compiled, not run on hardware
//
#include <hip/hip_runtime.h>
#include <hip/hip_bf16.h>

// ---------------------------------------------------------------------------
// MI455X (gfx1250) fused trajectory model.
// All GEMMs run as bf16 WMMA (v_wmma_f32_16x16x32_bf16, f32 accumulate).
// Weights are pre-converted to bf16 and stored N-major (transposed, K padded
// to a multiple of 32) so each B fragment is two contiguous 16B loads/lane.
// Activation tiles are DMA'd into LDS with GLOBAL_LOAD_ASYNC_TO_LDS_B128
// (ASYNCcnt path); each wave owns disjoint 16x16 output tiles (EXEC all ones
// around every WMMA).
// ---------------------------------------------------------------------------

typedef __bf16 bf16;
typedef __attribute__((ext_vector_type(16))) __bf16 v16bf;
typedef __attribute__((ext_vector_type(8)))  __bf16 v8bf;
typedef __attribute__((ext_vector_type(8)))  float  v8f;

#define Bc 2
#define Gc 4
#define Tc 16
#define Nc 500
#define Dd 256
#define DFFd 1024
#define Md 128
#define ROWS 32
#define NROWS_TOT (Bc * Gc * Tc * Nc)   // 64000
#define NBLK (NROWS_TOT / ROWS)         // 2000

// ---------------------------------------------------------------------- utils

__device__ inline float gelu_f(float x) {
  // jax.nn.gelu default (approximate=True, tanh form)
  float x3 = x * x * x;
  return 0.5f * x * (1.f + tanhf(0.7978845608028654f * (x + 0.044715f * x3)));
}

// Async DMA: 16 bytes per lane, global -> LDS, tracked by ASYNCcnt.
// LDS address = low 32 bits of the flat pointer (LDS aperture keeps the byte
// offset entirely in addr[31:0]).
__device__ inline void async_copy_b128(void* lds_dst, const void* gsrc) {
  unsigned ldsoff = (unsigned)(uintptr_t)lds_dst;
  asm volatile("global_load_async_to_lds_b128 %0, %1, off"
               :: "v"(ldsoff), "v"(gsrc)
               : "memory");
}
__device__ inline void async_wait_all() {
  asm volatile("s_wait_asynccnt 0" ::: "memory");
}

// A fragment: 16x32 bf16, A row-major in LDS with row pitch `pitch` elements.
// Lane l (0..31): row m0+(l&15); 16 contiguous K values at k0+16*(l>>4)
// (same K order as load_b_frag).
__device__ inline v16bf load_a_frag(const bf16* A, int pitch, int m0, int k0) {
  int l = threadIdx.x & 31;
  const bf16* p = A + (size_t)(m0 + (l & 15)) * pitch + k0 + ((l >> 4) << 4);
  union { v16bf v; v8bf h[2]; } u;
  u.h[0] = *(const v8bf*)p;        // ds_load_b128
  u.h[1] = *(const v8bf*)(p + 8);  // ds_load_b128
  return u.v;
}

// B fragment: 32x16 bf16 from N-major weight WT[n][Kp] in global memory.
// Lane l: column n0+(l&15); 16 contiguous K values at k0+16*(l>>4).
__device__ inline v16bf load_b_frag(const bf16* WT, int Kp, int n0, int k0) {
  int l = threadIdx.x & 31;
  const bf16* p = WT + (size_t)(n0 + (l & 15)) * Kp + k0 + ((l >> 4) << 4);
  union { v16bf v; v8bf h[2]; } u;
  u.h[0] = *(const v8bf*)p;        // global_load_b128
  u.h[1] = *(const v8bf*)(p + 8);  // global_load_b128
  return u.v;
}

__device__ inline v8f wmma_bf16(v16bf a, v16bf b, v8f c) {
  return __builtin_amdgcn_wmma_f32_16x16x32_bf16(
      /*neg_a=*/false, a, /*neg_b=*/false, b,
      /*c_mod=*/(short)0, c, /*reuse_a=*/false, /*reuse_b=*/false);
}

// ------------------------------------------------------- weight prep (bf16^T)

// WT[n*Kp + k] = (k < K) ? bf16(W[k*N + n]) : 0
__global__ void k_transpose_bf16(const float* __restrict__ W,
                                 bf16* __restrict__ WT, int K, int N, int Kp) {
  int idx = blockIdx.x * blockDim.x + threadIdx.x;
  if (idx >= N * Kp) return;
  int n = idx / Kp, k = idx - n * Kp;
  WT[idx] = (k < K) ? (bf16)W[(size_t)k * N + n] : (bf16)0.f;
}

// -------------------------------------------------- fused FFN (256->1024->256)

// Abuf: LDS [32][256] bf16; Hbuf: LDS [32][64] bf16 scratch.
// acc[4]: this wave's four 16x16 f32 output tiles (rt = w>>2, ct0 = (w&3)*4).
__device__ inline void ffn_block(const bf16* Abuf, bf16* Hbuf,
                                 const bf16* w1T, const float* b1,
                                 const bf16* w2T, v8f acc[4]) {
  const int w = threadIdx.x >> 5;
  const int l = threadIdx.x & 31;
  const int rt  = w >> 2;        // 0..1 (row tile)
  const int ct0 = (w & 3) * 4;   // output col-tile base (of 16)
  const int hct = w & 3;         // this wave's H col tile (of 4)
  const int rbase = rt * 16 + ((l >> 4) << 3);

  for (int c = 0; c < 16; ++c) {               // DFF in chunks of 64 columns
    if (c + 1 < 16)                            // global_prefetch_b8 next chunk
      __builtin_prefetch((const void*)(w1T + (size_t)((c + 1) * 64 + hct * 16) * Dd), 0, 1);
    // GEMM1: H = gelu(A @ w1[:, 64c:64c+64] + b1)
    v8f h = {};
#pragma unroll
    for (int kk = 0; kk < 8; ++kk) {
      v16bf a = load_a_frag(Abuf, Dd, rt * 16, kk * 32);
      v16bf b = load_b_frag(w1T, Dd, c * 64 + hct * 16, kk * 32);
      h = wmma_bf16(a, b, h);
    }
    __syncthreads();  // prior chunk's GEMM2 reads of Hbuf are done
    {
      int col = hct * 16 + (l & 15);
      float bias = b1[c * 64 + col];
#pragma unroll
      for (int v = 0; v < 8; ++v)
        Hbuf[(rbase + v) * 64 + col] = (bf16)gelu_f(h[v] + bias);
    }
    __syncthreads();
    // GEMM2: OUT += H @ w2[64c:64c+64, :]
#pragma unroll
    for (int j = 0; j < 4; ++j) {
      int ct = ct0 + j;
#pragma unroll
      for (int kk = 0; kk < 2; ++kk) {
        v16bf a = load_a_frag(Hbuf, 64, rt * 16, kk * 32);
        v16bf b = load_b_frag(w2T, DFFd, ct * 16, c * 64 + kk * 32);
        acc[j] = wmma_bf16(a, b, acc[j]);
      }
    }
  }
}

// ------------------------------------------------- stage A: x = LN(f + FFN(f))

__global__ __launch_bounds__(256) void k_stageA(
    const float* __restrict__ feats,
    const bf16* __restrict__ aw1T, const float* __restrict__ ab1,
    const bf16* __restrict__ aw2T, const float* __restrict__ ab2,
    const float* __restrict__ ng, const float* __restrict__ nb,
    bf16* __restrict__ xbf) {
  __shared__ __align__(16) unsigned char smem[65536];
  float* resid = (float*)smem;               // [32][256] f32   @0     (32KB)
  bf16*  Abuf  = (bf16*)(smem + 32768);      // [32][256] bf16  @32768 (16KB)
  bf16*  Hbuf  = (bf16*)(smem + 49152);      // [32][64]  bf16  @49152 (4KB)
  float* xres  = (float*)(smem + 32768);     // [32][256] f32 overlay (post-loop)

  const size_t base = (size_t)blockIdx.x * ROWS;
  const int tid = threadIdx.x;

  // feats tile (32KB, contiguous in both global and LDS) via async DMA.
  {
    const char* gsrc = (const char*)(feats + base * Dd);
    char* ldst = (char*)resid;
#pragma unroll
    for (int it = 0; it < 8; ++it)
      async_copy_b128(ldst + it * 4096 + tid * 16, gsrc + it * 4096 + tid * 16);
    async_wait_all();
  }
  __syncthreads();
  for (int i = tid; i < ROWS * Dd; i += 256) Abuf[i] = (bf16)resid[i];
  __syncthreads();

  v8f acc[4] = {};
  ffn_block(Abuf, Hbuf, aw1T, ab1, aw2T, acc);
  __syncthreads();  // Abuf/Hbuf dead -> xres may overlay

  {
    int w = tid >> 5, l = tid & 31;
    int rt = w >> 2, ct0 = (w & 3) * 4;
    int rbase = rt * 16 + ((l >> 4) << 3);
#pragma unroll
    for (int j = 0; j < 4; ++j) {
      int col = (ct0 + j) * 16 + (l & 15);
      float bias = ab2[col];
#pragma unroll
      for (int v = 0; v < 8; ++v) {
        int row = rbase + v;
        xres[row * Dd + col] = acc[j][v] + bias + resid[row * Dd + col];
      }
    }
  }
  __syncthreads();

  if (tid < ROWS) {
    float mu = 0.f;
    for (int n = 0; n < Dd; ++n) mu += xres[tid * Dd + n];
    mu *= (1.f / Dd);
    float var = 0.f;
    for (int n = 0; n < Dd; ++n) { float d = xres[tid * Dd + n] - mu; var += d * d; }
    var *= (1.f / Dd);
    float rinv = rsqrtf(var + 1e-5f);
    for (int n = 0; n < Dd; ++n) {
      float y = (xres[tid * Dd + n] - mu) * rinv * ng[n] + nb[n];
      xbf[(base + tid) * Dd + n] = (bf16)y;
    }
  }
}

// ------------------------------------------------ stage M: motion MLP -> m

__global__ __launch_bounds__(256) void k_motion(
    const bf16* __restrict__ xbf, const float* __restrict__ boxes,
    const unsigned char* __restrict__ masks,
    const bf16* __restrict__ mw1T, const float* __restrict__ mb1,
    const bf16* __restrict__ mw2T, const float* __restrict__ mb2,
    const bf16* __restrict__ mw3T, const float* __restrict__ mb3,
    bf16* __restrict__ mbf) {
  __shared__ __align__(16) unsigned char smem[34944];
  bf16*  A     = (bf16*)smem;                // [32][288] bf16 (K=264 padded)
  bf16*  H1    = (bf16*)(smem + 18432);      // [32][128] bf16
  bf16*  H2    = (bf16*)(smem + 26624);      // [32][128] bf16
  float* maskf = (float*)(smem + 34816);     // [32] f32

  const size_t base = (size_t)blockIdx.x * ROWS;
  const int tid = threadIdx.x;

  // x tile: 512B/row -> LDS rows at pitch 576B, via async DMA (4 issues).
  {
    const char* xsrc = (const char*)xbf + base * Dd * 2;
    char* adst = (char*)A;
    int r0 = tid >> 5, n16 = tid & 31;
#pragma unroll
    for (int it = 0; it < 4; ++it) {
      int r = r0 + it * 8;
      async_copy_b128(adst + (size_t)r * 576 + n16 * 16,
                      xsrc + (size_t)r * 512 + n16 * 16);
    }
    async_wait_all();
  }
  if (tid < ROWS) {
    int r = tid;
    size_t grow = base + r;
    int t = (int)((grow / Nc) % Tc);
    size_t prev = (t > 0) ? (grow - Nc) : grow;
#pragma unroll
    for (int q = 0; q < 4; ++q) {
      float bq = boxes[grow * 4 + q];
      float pq = boxes[prev * 4 + q];
      A[r * 288 + 256 + q] = (bf16)bq;
      A[r * 288 + 260 + q] = (bf16)(bq - pq);   // vel (0 at t==0)
    }
    for (int q = 264; q < 288; ++q) A[r * 288 + q] = (bf16)0.f;  // K padding
    maskf[r] = masks[grow] ? 1.f : 0.f;
  }
  __syncthreads();

  const int w = tid >> 5, l = tid & 31;
  const int rt = w >> 2, ct2 = (w & 3) * 2;   // 2 of 8 col tiles per wave
  const int rbase = rt * 16 + ((l >> 4) << 3);

  // layer 1: gelu([x,b,v] @ mw1 + mb1), K=288
#pragma unroll
  for (int j = 0; j < 2; ++j) {
    int ct = ct2 + j;
    v8f acc = {};
#pragma unroll
    for (int kk = 0; kk < 9; ++kk) {
      v16bf a = load_a_frag(A, 288, rt * 16, kk * 32);
      v16bf b = load_b_frag(mw1T, 288, ct * 16, kk * 32);
      acc = wmma_bf16(a, b, acc);
    }
    int col = ct * 16 + (l & 15);
    float bias = mb1[col];
#pragma unroll
    for (int v = 0; v < 8; ++v)
      H1[(rbase + v) * Md + col] = (bf16)gelu_f(acc[v] + bias);
  }
  __syncthreads();

  // layer 2: gelu(H1 @ mw2 + mb2), K=128
#pragma unroll
  for (int j = 0; j < 2; ++j) {
    int ct = ct2 + j;
    v8f acc = {};
#pragma unroll
    for (int kk = 0; kk < 4; ++kk) {
      v16bf a = load_a_frag(H1, Md, rt * 16, kk * 32);
      v16bf b = load_b_frag(mw2T, Md, ct * 16, kk * 32);
      acc = wmma_bf16(a, b, acc);
    }
    int col = ct * 16 + (l & 15);
    float bias = mb2[col];
#pragma unroll
    for (int v = 0; v < 8; ++v)
      H2[(rbase + v) * Md + col] = (bf16)gelu_f(acc[v] + bias);
  }
  __syncthreads();

  // layer 3: (H2 @ mw3 + mb3) * mask  -> mbf (bf16)
#pragma unroll
  for (int j = 0; j < 2; ++j) {
    int ct = ct2 + j;
    v8f acc = {};
#pragma unroll
    for (int kk = 0; kk < 4; ++kk) {
      v16bf a = load_a_frag(H2, Md, rt * 16, kk * 32);
      v16bf b = load_b_frag(mw3T, Md, ct * 16, kk * 32);
      acc = wmma_bf16(a, b, acc);
    }
    int col = ct * 16 + (l & 15);
    float bias = mb3[col];
#pragma unroll
    for (int v = 0; v < 8; ++v) {
      int row = rbase + v;
      mbf[(base + row) * Md + col] = (bf16)((acc[v] + bias) * maskf[row]);
    }
  }
}

// --------------------------------------------------- Kalman (closed form)

// innov = z - xp[:4] == 0 because xp[:4] = z + 0 -> state = [z, 0] always;
// cov = upd ? Pn : I8, with Pn a constant 2x2-block matrix (blocks * I4).
__global__ void k_kalman(const float* __restrict__ boxes,
                         const unsigned char* __restrict__ masks,
                         float* __restrict__ kstate, float* __restrict__ kcov) {
  int tr = blockIdx.x * blockDim.x + threadIdx.x;
  if (tr >= Bc * Gc * Nc) return;
  int n = tr % Nc;
  int bg = tr / Nc;
  size_t row = ((size_t)bg * Tc + (Tc - 1)) * Nc + n;   // last frame
  bool upd = (masks[row] == 0);
#pragma unroll
  for (int q = 0; q < 4; ++q) {
    kstate[(size_t)tr * 8 + q] = boxes[row * 4 + q];
    kstate[(size_t)tr * 8 + 4 + q] = 0.f;
  }
  const float a = 2.01f, b = 1.0f, c = 1.01f, s = 2.11f;  // Pp blocks, S
  float p00 = upd ? (a - a * a / s) : 1.f;
  float p01 = upd ? (b - a * b / s) : 0.f;
  float p11 = upd ? (c - b * b / s) : 1.f;
  for (int i = 0; i < 8; ++i)
    for (int j = 0; j < 8; ++j) {
      float v = 0.f;
      if ((i & 3) == (j & 3)) {
        int bi = i >> 2, bj = j >> 2;
        v = (bi == 0 && bj == 0) ? p00 : ((bi == 1 && bj == 1) ? p11 : p01);
      }
      kcov[(size_t)tr * 64 + i * 8 + j] = v;
    }
}

// ------------------ stage F: fusion MLP + FFN2 + final LN -> d_out (f32)

__global__ __launch_bounds__(256) void k_stageF(
    const bf16* __restrict__ xbf, const bf16* __restrict__ mbf,
    const bf16* __restrict__ fw1T, const float* __restrict__ fb1,
    const bf16* __restrict__ fw2T, const float* __restrict__ fb2,
    const bf16* __restrict__ w1T, const float* __restrict__ b1,
    const bf16* __restrict__ w2T, const float* __restrict__ b2,
    const float* __restrict__ fng, const float* __restrict__ fnb,
    float* __restrict__ out) {
  __shared__ __align__(16) unsigned char smem[65536];
  bf16*  T1   = (bf16*)smem;                 // [32][256] bf16 @0     (dead after G2)
  bf16*  Fbuf = (bf16*)(smem + 16384);       // [32][384] bf16 @16384 (dead after G1)
  float* X2   = (float*)(smem + 16384);      // [32][256] f32 overlay (from G2)
  bf16*  A2   = (bf16*)(smem + 49152);       // [32][256] bf16 @49152
  bf16*  Hbuf = (bf16*)smem;                 // [32][64]  bf16 overlay on T1

  const size_t base = (size_t)blockIdx.x * ROWS;
  const int tid = threadIdx.x;

  // [x | m] tile -> Fbuf (pitch 768B) via async DMA (4 + 2 issues).
  {
    const char* xsrc = (const char*)xbf + base * Dd * 2;
    const char* msrc = (const char*)mbf + base * Md * 2;
    char* fdst = (char*)Fbuf;
    int r0 = tid >> 5, n16 = tid & 31;
#pragma unroll
    for (int it = 0; it < 4; ++it) {
      int r = r0 + it * 8;
      async_copy_b128(fdst + (size_t)r * 768 + n16 * 16,
                      xsrc + (size_t)r * 512 + n16 * 16);
    }
    int r1 = tid >> 4, n8 = tid & 15;
#pragma unroll
    for (int it = 0; it < 2; ++it) {
      int r = r1 + it * 16;
      async_copy_b128(fdst + (size_t)r * 768 + 512 + n8 * 16,
                      msrc + (size_t)r * 256 + n8 * 16);
    }
    async_wait_all();
  }
  __syncthreads();

  const int w = tid >> 5, l = tid & 31;
  const int rt = w >> 2, ct0 = (w & 3) * 4;
  const int rbase = rt * 16 + ((l >> 4) << 3);

  // G1: T1 = relu([x,m] @ fw1 + fb1), K=384
#pragma unroll
  for (int j = 0; j < 4; ++j) {
    int ct = ct0 + j;
    v8f acc = {};
#pragma unroll
    for (int kk = 0; kk < 12; ++kk) {
      v16bf a = load_a_frag(Fbuf, 384, rt * 16, kk * 32);
      v16bf b = load_b_frag(fw1T, 384, ct * 16, kk * 32);
      acc = wmma_bf16(a, b, acc);
    }
    int col = ct * 16 + (l & 15);
    float bias = fb1[col];
#pragma unroll
    for (int v = 0; v < 8; ++v)
      T1[(rbase + v) * Dd + col] = (bf16)fmaxf(acc[v] + bias, 0.f);
  }
  __syncthreads();  // Fbuf dead -> X2/A2 regions usable

  // G2: x2 = T1 @ fw2 + fb2 -> X2 (f32 residual) and A2 (bf16 for FFN2)
#pragma unroll
  for (int j = 0; j < 4; ++j) {
    int ct = ct0 + j;
    v8f acc = {};
#pragma unroll
    for (int kk = 0; kk < 8; ++kk) {
      v16bf a = load_a_frag(T1, Dd, rt * 16, kk * 32);
      v16bf b = load_b_frag(fw2T, Dd, ct * 16, kk * 32);
      acc = wmma_bf16(a, b, acc);
    }
    int col = ct * 16 + (l & 15);
    float bias = fb2[col];
#pragma unroll
    for (int v = 0; v < 8; ++v) {
      float x2 = acc[v] + bias;
      X2[(rbase + v) * Dd + col] = x2;
      A2[(rbase + v) * Dd + col] = (bf16)x2;
    }
  }
  __syncthreads();  // T1 dead -> Hbuf overlay usable

  v8f acc[4] = {};
  ffn_block(A2, Hbuf, w1T, b1, w2T, acc);
  __syncthreads();

#pragma unroll
  for (int j = 0; j < 4; ++j) {
    int col = (ct0 + j) * 16 + (l & 15);
    float bias = b2[col];
#pragma unroll
    for (int v = 0; v < 8; ++v) {
      int row = rbase + v;
      X2[row * Dd + col] += acc[j][v] + bias;   // x2 + FFN2(x2)
    }
  }
  __syncthreads();

  if (tid < ROWS) {
    float mu = 0.f;
    for (int n = 0; n < Dd; ++n) mu += X2[tid * Dd + n];
    mu *= (1.f / Dd);
    float var = 0.f;
    for (int n = 0; n < Dd; ++n) { float d = X2[tid * Dd + n] - mu; var += d * d; }
    var *= (1.f / Dd);
    float rinv = rsqrtf(var + 1e-5f);
    for (int n = 0; n < Dd; ++n)
      out[(base + tid) * Dd + n] = (X2[tid * Dd + n] - mu) * rinv * fng[n] + fnb[n];
  }
}

// ------------------------------------------------------------------- launcher

extern "C" void kernel_launch(void* const* d_in, const int* in_sizes, int n_in,
                              void* d_out, int out_size, void* d_ws, size_t ws_size,
                              hipStream_t stream) {
  (void)in_sizes; (void)n_in; (void)out_size; (void)ws_size;
  const float* feats = (const float*)d_in[0];
  const float* boxes = (const float*)d_in[1];
  const unsigned char* masks = (const unsigned char*)d_in[2];
  const float* aw1 = (const float*)d_in[3];
  const float* ab1 = (const float*)d_in[4];
  const float* aw2 = (const float*)d_in[5];
  const float* ab2 = (const float*)d_in[6];
  const float* ng  = (const float*)d_in[7];
  const float* nb  = (const float*)d_in[8];
  const float* mw1 = (const float*)d_in[9];
  const float* mb1 = (const float*)d_in[10];
  const float* mw2 = (const float*)d_in[11];
  const float* mb2 = (const float*)d_in[12];
  const float* mw3 = (const float*)d_in[13];
  const float* mb3 = (const float*)d_in[14];
  const float* fw1 = (const float*)d_in[15];
  const float* fb1 = (const float*)d_in[16];
  const float* fw2 = (const float*)d_in[17];
  const float* fb2 = (const float*)d_in[18];
  const float* w1  = (const float*)d_in[19];
  const float* b1  = (const float*)d_in[20];
  const float* w2  = (const float*)d_in[21];
  const float* b2  = (const float*)d_in[22];
  const float* fng = (const float*)d_in[23];
  const float* fnb = (const float*)d_in[24];

  char* ws = (char*)d_ws;
  size_t off = 0;
  auto wsalloc = [&](size_t bytes) -> void* {
    void* p = ws + off;
    off += (bytes + 255) & ~(size_t)255;
    return p;
  };
  bf16* aw1T = (bf16*)wsalloc((size_t)DFFd * Dd * 2);     // [1024][256]
  bf16* aw2T = (bf16*)wsalloc((size_t)Dd * DFFd * 2);     // [256][1024]
  bf16* w1T  = (bf16*)wsalloc((size_t)DFFd * Dd * 2);
  bf16* w2T  = (bf16*)wsalloc((size_t)Dd * DFFd * 2);
  bf16* mw1T = (bf16*)wsalloc((size_t)Md * 288 * 2);      // [128][288] padded
  bf16* mw2T = (bf16*)wsalloc((size_t)Md * Md * 2);
  bf16* mw3T = (bf16*)wsalloc((size_t)Md * Md * 2);
  bf16* fw1T = (bf16*)wsalloc((size_t)Dd * 384 * 2);      // [256][384]
  bf16* fw2T = (bf16*)wsalloc((size_t)Dd * Dd * 2);
  bf16* xbf  = (bf16*)wsalloc((size_t)NROWS_TOT * Dd * 2);
  bf16* mbf  = (bf16*)wsalloc((size_t)NROWS_TOT * Md * 2);

  auto tl = [&](const float* W, bf16* WT, int K, int N, int Kp) {
    int total = N * Kp;
    k_transpose_bf16<<<(total + 255) / 256, 256, 0, stream>>>(W, WT, K, N, Kp);
  };
  tl(aw1, aw1T, Dd, DFFd, Dd);     // 256x1024 -> [1024][256]
  tl(aw2, aw2T, DFFd, Dd, DFFd);   // 1024x256 -> [256][1024]
  tl(w1,  w1T,  Dd, DFFd, Dd);
  tl(w2,  w2T,  DFFd, Dd, DFFd);
  tl(mw1, mw1T, Dd + 8, Md, 288);  // 264x128 -> [128][288] zero-padded
  tl(mw2, mw2T, Md, Md, Md);
  tl(mw3, mw3T, Md, Md, Md);
  tl(fw1, fw1T, Dd + Md, Dd, 384); // 384x256 -> [256][384]
  tl(fw2, fw2T, Dd, Dd, Dd);

  k_stageA<<<NBLK, 256, 0, stream>>>(feats, aw1T, ab1, aw2T, ab2, ng, nb, xbf);
  k_motion<<<NBLK, 256, 0, stream>>>(xbf, boxes, masks, mw1T, mb1, mw2T, mb2,
                                     mw3T, mb3, mbf);

  float* out = (float*)d_out;
  float* kstate = out + (size_t)NROWS_TOT * Dd;              // 16,384,000
  float* kcov   = kstate + (size_t)Bc * Gc * Nc * 8;         // +32,000
  k_kalman<<<(Bc * Gc * Nc + 255) / 256, 256, 0, stream>>>(boxes, masks, kstate, kcov);

  k_stageF<<<NBLK, 256, 0, stream>>>(xbf, mbf, fw1T, fb1, fw2T, fb2,
                                     w1T, b1, w2T, b2, fng, fnb, out);
}
